// Encoder_Learner_43327630082687
// MI455X (gfx1250) — matrix operand order repaired and checked by reading, not compiled, not run on hardware
//
#include <hip/hip_runtime.h>
#include <hip/hip_bf16.h>
#include <math.h>

typedef __attribute__((ext_vector_type(16))) __bf16 v16bf;
typedef __attribute__((ext_vector_type(8)))  float  v8f;

__device__ inline v8f wmma_bf16(v16bf a, v16bf b, v8f c) {
#if defined(__HIP_DEVICE_COMPILE__)
  // 8 args: (neg_a, A, neg_b, B, c_mod, C, reuse_a, reuse_b)
  return __builtin_amdgcn_wmma_f32_16x16x32_bf16(false, a, false, b, (short)0, c,
                                                 false, false);
#else
  (void)a; (void)b;
  return c;
#endif
}

// ---------------------------------------------------------------------------
// Kernel 1: pack f32 weight [K][N] row-major into WMMA bf16 B-operand layout.
// Chunk per (kstep, ntile): 32 lanes x 16 bf16 (contiguous per lane).
// B k-order used here (contiguous k, 32x16): lanes 0-15 hold col n, K=0..15; lanes 16-31 hold
// col n, K=16..31 (pairs packed per dword sequentially).
// ---------------------------------------------------------------------------
__global__ __launch_bounds__(256)
void pack_w_kernel(const float* __restrict__ W, __bf16* __restrict__ out,
                   int N, int total) {
  int tid = blockIdx.x * 256 + threadIdx.x;
  if (tid >= total) return;
  int j    = tid & 15;          // element within lane
  int lane = (tid >> 4) & 31;
  int rest = tid >> 9;          // kstep * NT + ntile
  int NT = N >> 4;
  int nt = rest % NT;
  int ks = rest / NT;
  int kk = j + ((lane >> 4) << 4);       // K within 32-step
  int n  = nt * 16 + (lane & 15);
  out[tid] = (__bf16)W[(size_t)(ks * 32 + kk) * N + n];
}

// ---------------------------------------------------------------------------
// Kernel 2: y = (feat_slice @ W) / max(||row||,eps) + bias   (bf16 WMMA)
// One block = 16 rows (M-tile), 128 threads = 4 waves, wave w owns N-tiles
// {3w, 3w+1, 3w+2} of the 12 tiles (N = 192). Row L2-norm accumulated while
// staging the f32->bf16 A-tile through LDS in contiguous-k order (k = 16h + s, as packed B).
// ---------------------------------------------------------------------------
__global__ __launch_bounds__(128)
void gemm_rownorm_kernel(const float* __restrict__ feat, int ldf, int kStart,
                         int K, const __bf16* __restrict__ Bp,
                         const float* __restrict__ bias,
                         float* __restrict__ yOut) {
  __shared__ __align__(32) __bf16 ldsA[32][16];
  __shared__ float ldsNorm[16];
  const int t     = threadIdx.x;
  const int mBase = blockIdx.x * 16;
  const int row   = t >> 3;            // 0..15, 8 threads per row
  const int k0    = (t & 7) * 4;       // quad of K within 32-step
  const int blk   = k0 >> 3;           // which 8-K block (0..3)
  // A order used here (contiguous k): lanes 0-15 = row, K{0-15}; lanes 16-31 = K{16-31}
  const int aLane = row + ((blk >> 1) << 4);
  const int aSlot = (k0 & 7) + ((blk & 1) << 3);
  const int lane  = t & 31;
  const int w     = t >> 5;
  const int half8 = (lane >> 4) << 3;  // C layout: vgpr i -> row i (+8 upper half)
  const float* src = feat + (size_t)(mBase + row) * ldf + kStart + k0;

  v8f acc[3] = {};
  float sq = 0.f;
  const int ksteps = K >> 5;
  for (int ks = 0; ks < ksteps; ++ks) {
    float4 f = *(const float4*)(src + ks * 32);
    if (ks + 1 < ksteps) __builtin_prefetch(src + (ks + 1) * 32, 0, 1);
    sq = fmaf(f.x, f.x, sq); sq = fmaf(f.y, f.y, sq);
    sq = fmaf(f.z, f.z, sq); sq = fmaf(f.w, f.w, sq);
    ldsA[aLane][aSlot + 0] = (__bf16)f.x;
    ldsA[aLane][aSlot + 1] = (__bf16)f.y;
    ldsA[aLane][aSlot + 2] = (__bf16)f.z;
    ldsA[aLane][aSlot + 3] = (__bf16)f.w;
    __syncthreads();
    v16bf a = *(const v16bf*)(&ldsA[lane][0]);
    const __bf16* bp = Bp + (((size_t)ks * 12 + w * 3) << 9) + (lane << 4);
#pragma unroll
    for (int j = 0; j < 3; ++j) {
      v16bf bm = *(const v16bf*)(bp + ((size_t)j << 9));
      acc[j] = wmma_bf16(a, bm, acc[j]);
    }
    __syncthreads();
  }
  // reduce row sum-of-squares across the 8 threads of each row
  sq += __shfl_xor(sq, 1);
  sq += __shfl_xor(sq, 2);
  sq += __shfl_xor(sq, 4);
  if ((t & 7) == 0) ldsNorm[row] = sqrtf(sq);
  __syncthreads();
#pragma unroll
  for (int j = 0; j < 3; ++j) {
    const int col = (w * 3 + j) * 16 + (lane & 15);
    const float bc = bias[col];
#pragma unroll
    for (int i = 0; i < 8; ++i) {
      const int r = i + half8;
      const float rn = 1.f / fmaxf(ldsNorm[r], 1e-12f);
      yOut[(size_t)(mBase + r) * 192 + col] = fmaf(acc[j][i], rn, bc);
    }
  }
}

// ---------------------------------------------------------------------------
// Kernel 3: per-channel batch stats -> affine (scale, shift) for BN.
// One block per channel (192 blocks per modality).
// ---------------------------------------------------------------------------
__global__ __launch_bounds__(256)
void bn_stats_kernel(const float* __restrict__ y, const float* __restrict__ gamma,
                     const float* __restrict__ beta, float* __restrict__ scale,
                     float* __restrict__ shift, int nItems) {
  __shared__ float ls[256], ls2[256];
  const int c = blockIdx.x;
  float s = 0.f, s2 = 0.f;
  for (int i = threadIdx.x; i < nItems; i += 256) {
    float v = y[(size_t)i * 192 + c];
    s += v;
    s2 = fmaf(v, v, s2);
  }
  ls[threadIdx.x] = s; ls2[threadIdx.x] = s2;
  __syncthreads();
  for (int off = 128; off > 0; off >>= 1) {
    if (threadIdx.x < off) {
      ls[threadIdx.x]  += ls[threadIdx.x + off];
      ls2[threadIdx.x] += ls2[threadIdx.x + off];
    }
    __syncthreads();
  }
  if (threadIdx.x == 0) {
    float inv = 1.f / (float)nItems;
    float mean = ls[0] * inv;
    float var  = ls2[0] * inv - mean * mean;
    float sc   = gamma[c] * rsqrtf(var + 1e-5f);
    scale[c] = sc;
    shift[c] = beta[c] - mean * sc;
  }
}

// ---------------------------------------------------------------------------
// Kernel 4: h = 0.5*relu(bn(yv)) + 0.5*relu(bn(yt)); feats = h @ f_W + f_b.
// 16-row M-tiles, 4 waves, wave w owns N-tile w (N = 64). Writes raw feats
// (for scatter) and L2-normalized feats (output region 2).
// ---------------------------------------------------------------------------
__global__ __launch_bounds__(128)
void fuse_gemm_kernel(const float* __restrict__ yv, const float* __restrict__ yt,
                      const float* __restrict__ vs, const float* __restrict__ vh,
                      const float* __restrict__ tsc, const float* __restrict__ tsh,
                      const __bf16* __restrict__ Bf, const float* __restrict__ fb,
                      float* __restrict__ rawOut, float* __restrict__ normOut) {
  __shared__ __align__(32) __bf16 ldsH[32][16];
  __shared__ float rowSq[16];
  const int t = threadIdx.x;
  if (t < 16) rowSq[t] = 0.f;
  const int mBase = blockIdx.x * 16;
  const int row = t >> 3;
  const int k0  = (t & 7) * 4;
  const int blk = k0 >> 3;
  const int aLane = row + ((blk >> 1) << 4);
  const int aSlot = (k0 & 7) + ((blk & 1) << 3);
  const int lane = t & 31;
  const int w    = t >> 5;
  const int half8 = (lane >> 4) << 3;

  v8f acc = {};
  for (int ks = 0; ks < 6; ++ks) {          // K = 192 = 6 * 32
    const int c0 = ks * 32 + k0;
    float4 a4 = *(const float4*)(yv + (size_t)(mBase + row) * 192 + c0);
    float4 b4 = *(const float4*)(yt + (size_t)(mBase + row) * 192 + c0);
    float av[4] = {a4.x, a4.y, a4.z, a4.w};
    float bv[4] = {b4.x, b4.y, b4.z, b4.w};
#pragma unroll
    for (int i = 0; i < 4; ++i) {
      int c = c0 + i;
      float x = fmaxf(fmaf(av[i], vs[c],  vh[c]),  0.f);
      float y = fmaxf(fmaf(bv[i], tsc[c], tsh[c]), 0.f);
      ldsH[aLane][aSlot + i] = (__bf16)(0.5f * x + 0.5f * y);  // V_W = T_W = 0.5
    }
    __syncthreads();
    v16bf a = *(const v16bf*)(&ldsH[lane][0]);
    const __bf16* bp = Bf + (((size_t)ks * 4 + w) << 9) + (lane << 4);
    v16bf bm = *(const v16bf*)bp;
    acc = wmma_bf16(a, bm, acc);
    __syncthreads();
  }
  const int col = w * 16 + (lane & 15);
  const float fbias = fb[col];
  float vals[8];
#pragma unroll
  for (int i = 0; i < 8; ++i) {
    vals[i] = acc[i] + fbias;
    float s = vals[i] * vals[i];
    s += __shfl_xor(s, 1);
    s += __shfl_xor(s, 2);
    s += __shfl_xor(s, 4);
    s += __shfl_xor(s, 8);
    if ((lane & 15) == 0) atomicAdd(&rowSq[i + half8], s);
  }
  __syncthreads();
#pragma unroll
  for (int i = 0; i < 8; ++i) {
    const int r = i + half8;
    const float rn = 1.f / fmaxf(sqrtf(rowSq[r]), 1e-12f);
    const size_t o = (size_t)(mBase + r) * 64 + col;
    rawOut[o]  = vals[i];
    normOut[o] = vals[i] * rn;
  }
}

// ---------------------------------------------------------------------------
// Kernel 5: zero a float4 region (user accumulator must be zeroed every call).
// ---------------------------------------------------------------------------
__global__ __launch_bounds__(256)
void zero_kernel(float4* __restrict__ p, size_t n4) {
  size_t i = (size_t)blockIdx.x * 256 + threadIdx.x;
  if (i < n4) p[i] = make_float4(0.f, 0.f, 0.f, 0.f);
}

// ---------------------------------------------------------------------------
// Kernel 6: segment-sum scatter over edges. 16 lanes per edge, float4 each.
// ---------------------------------------------------------------------------
__global__ __launch_bounds__(256)
void scatter_kernel(const int* __restrict__ eu, const int* __restrict__ ei,
                    const float* __restrict__ raw, float* __restrict__ uv,
                    int nnz) {
  int e = blockIdx.x * 16 + (threadIdx.x >> 4);
  int l = threadIdx.x & 15;
  if (e >= nnz) return;
  int u  = eu[e];
  int it = ei[e];
  float4 v = *(const float4*)(raw + (size_t)it * 64 + l * 4);
  float* dst = uv + (size_t)u * 64 + l * 4;
  atomicAdd(dst + 0, v.x);
  atomicAdd(dst + 1, v.y);
  atomicAdd(dst + 2, v.z);
  atomicAdd(dst + 3, v.w);
}

// ---------------------------------------------------------------------------
// Kernel 7: row-wise L2 normalize [rows, 64]. 16 lanes per row.
// ---------------------------------------------------------------------------
__global__ __launch_bounds__(256)
void norm_rows_kernel(const float* __restrict__ in, float* __restrict__ out,
                      int rows) {
  int r = (blockIdx.x * 256 + threadIdx.x) >> 4;
  int l = threadIdx.x & 15;
  if (r >= rows) return;
  float4 v = *(const float4*)(in + (size_t)r * 64 + l * 4);
  float s = v.x * v.x + v.y * v.y + v.z * v.z + v.w * v.w;
  s += __shfl_xor(s, 1);
  s += __shfl_xor(s, 2);
  s += __shfl_xor(s, 4);
  s += __shfl_xor(s, 8);
  float rn = 1.f / fmaxf(sqrtf(s), 1e-12f);
  float* o = out + (size_t)r * 64 + l * 4;
  o[0] = v.x * rn; o[1] = v.y * rn; o[2] = v.z * rn; o[3] = v.w * rn;
}

// ---------------------------------------------------------------------------
extern "C" void kernel_launch(void* const* d_in, const int* in_sizes, int n_in,
                              void* d_out, int out_size, void* d_ws, size_t ws_size,
                              hipStream_t stream) {
  const float* feat      = (const float*)d_in[0];
  const int*   edge_user = (const int*)d_in[1];
  const int*   edge_item = (const int*)d_in[2];
  const float* v_W  = (const float*)d_in[3];
  const float* v_b  = (const float*)d_in[4];
  const float* v_g  = (const float*)d_in[5];
  const float* v_be = (const float*)d_in[6];
  const float* t_W  = (const float*)d_in[7];
  const float* t_b  = (const float*)d_in[8];
  const float* t_g  = (const float*)d_in[9];
  const float* t_be = (const float*)d_in[10];
  const float* f_W  = (const float*)d_in[11];
  const float* f_b  = (const float*)d_in[12];

  const int n_items = in_sizes[0] / 4480;
  const int nnz     = in_sizes[1];
  const int n_users = out_size / 64 - n_items;

  // ---- workspace partition (256 B aligned) ----
  char* ws = (char*)d_ws;
  size_t off = 0;
  auto carve = [&](size_t bytes) {
    void* p = ws + off;
    off += (bytes + 255) & ~(size_t)255;
    return p;
  };
  const int pbv_total = (4096 / 32) * (192 / 16) * 512;   // 786432
  const int pbt_total = (384  / 32) * (192 / 16) * 512;   //  73728
  const int pbf_total = (192  / 32) * (64  / 16) * 512;   //  12288
  __bf16* pbv = (__bf16*)carve((size_t)pbv_total * sizeof(__bf16));
  __bf16* pbt = (__bf16*)carve((size_t)pbt_total * sizeof(__bf16));
  __bf16* pbf = (__bf16*)carve((size_t)pbf_total * sizeof(__bf16));
  float* yv     = (float*)carve((size_t)n_items * 192 * sizeof(float));
  float* yt     = (float*)carve((size_t)n_items * 192 * sizeof(float));
  float* vscale = (float*)carve(192 * sizeof(float));
  float* vshift = (float*)carve(192 * sizeof(float));
  float* tscale = (float*)carve(192 * sizeof(float));
  float* tshift = (float*)carve(192 * sizeof(float));
  float* raw    = (float*)carve((size_t)n_items * 64 * sizeof(float));
  float* uvec   = (float*)carve((size_t)n_users * 64 * sizeof(float));
  (void)ws_size; (void)n_in;

  float* outUsers = (float*)d_out;                       // [n_users, 64]
  float* outItems = (float*)d_out + (size_t)n_users * 64; // [n_items, 64]

  // 1) pack weights into bf16 WMMA B layout
  pack_w_kernel<<<(pbv_total + 255) / 256, 256, 0, stream>>>(v_W, pbv, 192, pbv_total);
  pack_w_kernel<<<(pbt_total + 255) / 256, 256, 0, stream>>>(t_W, pbt, 192, pbt_total);
  pack_w_kernel<<<(pbf_total + 255) / 256, 256, 0, stream>>>(f_W, pbf, 64,  pbf_total);

  // 5) zero user accumulator (independent; overlaps with GEMMs)
  const size_t uv4 = (size_t)n_users * 16;
  zero_kernel<<<(unsigned)((uv4 + 255) / 256), 256, 0, stream>>>((float4*)uvec, uv4);

  // 2) modality GEMMs with fused row L2-norm + bias
  const int mTiles = n_items / 16;  // 50000 / 16 = 3125 exact
  gemm_rownorm_kernel<<<mTiles, 128, 0, stream>>>(feat, 4480, 0,    4096, pbv, v_b, yv);
  gemm_rownorm_kernel<<<mTiles, 128, 0, stream>>>(feat, 4480, 4096, 384,  pbt, t_b, yt);

  // 3) batch-norm statistics -> affine
  bn_stats_kernel<<<192, 256, 0, stream>>>(yv, v_g, v_be, vscale, vshift, n_items);
  bn_stats_kernel<<<192, 256, 0, stream>>>(yt, t_g, t_be, tscale, tshift, n_items);

  // 4) fused BN-ReLU-mix + final GEMM (+ item L2 norm -> output 2)
  fuse_gemm_kernel<<<mTiles, 128, 0, stream>>>(yv, yt, vscale, vshift, tscale, tshift,
                                               pbf, f_b, raw, outItems);

  // 6) edge scatter (segment sum)
  scatter_kernel<<<(nnz + 15) / 16, 256, 0, stream>>>(edge_user, edge_item, raw, uvec, nnz);

  // 7) user L2 norm -> output 1
  norm_rows_kernel<<<(n_users * 16 + 255) / 256, 256, 0, stream>>>(uvec, outUsers, n_users);
}
